// ODEFunc_29901562314988
// MI455X (gfx1250) — matrix operand order repaired and checked by reading, not compiled, not run on hardware
//
#include <hip/hip_runtime.h>
#include <hip/hip_bf16.h>

// ---------------------------------------------------------------------------
// CNF ODE func on MI455X (gfx1250, wave32, WMMA).
// dz_dt = MLP([z,t]);  dlogp_dt = -tr(J) = -(d2^T K d1),  K = W2 .* (W1z W3)^T
// ---------------------------------------------------------------------------

typedef __attribute__((ext_vector_type(16))) __bf16        v16bf;
typedef __attribute__((ext_vector_type(8)))  float         v8f;
typedef __attribute__((ext_vector_type(4)))  unsigned int  v4u;

#define NB   8192
#define DIMZ 64
#define HD   256
#define ZP   72    // padded LDS row stride (ushort) for z tile
#define HP   264   // padded LDS row stride (ushort) for H=256 activations
#define FP   260   // padded LDS row stride (float)  for d2 / u

union V16 { v4u q[2]; v16bf v; };

__device__ __forceinline__ unsigned short f2bf(float f) {
  union { float f; unsigned u; } c; c.f = f;
  unsigned r = c.u + 0x7FFFu + ((c.u >> 16) & 1u);   // round-to-nearest-even
  return (unsigned short)(r >> 16);
}

// fast sigmoid: v_exp_f32 + v_rcp_f32 (no IEEE division sequence)
__device__ __forceinline__ float fsigmoid(float x) {
  return __builtin_amdgcn_rcpf(1.f + __expf(-x));
}

__device__ __forceinline__ v8f wmma_bf16(v16bf a, v16bf b, v8f c) {
  // emits v_wmma_f32_16x16x32_bf16
  return __builtin_amdgcn_wmma_f32_16x16x32_bf16(false, a, false, b,
                                                 (short)0, c, false, false);
}

// A operand (16x32 bf16) gathered from row-major bf16 LDS buffer.
// ISA layout: lane 0-15 row m=lane, K in {0..7}U{16..23}; lanes 16-31 K+8.
// Both 8-element runs are contiguous -> two 16B LDS loads.
__device__ __forceinline__ v16bf load_a(const unsigned short* buf, int stride,
                                        int kt, int lane) {
  int m = lane & 15;
  const unsigned short* p = buf + m * stride + (kt << 5) + ((lane >> 4) << 3);
  V16 t;
  t.q[0] = *reinterpret_cast<const v4u*>(p);
  t.q[1] = *reinterpret_cast<const v4u*>(p + 16);
  return t.v;
}

// B operand from pre-packed global buffer: tile = 32 lanes x 16 bf16 (1KB),
// each lane reads its 32 contiguous bytes (two b128 loads).
__device__ __forceinline__ v16bf load_b(const unsigned short* pack, int tile,
                                        int lane) {
  const v4u* p = reinterpret_cast<const v4u*>(pack + ((size_t)tile << 9) + (lane << 4));
  V16 t; t.q[0] = p[0]; t.q[1] = p[1];
  return t.v;
}

// ------------------------------ prep kernels -------------------------------

// K[i,j] = W2[i,j] * M[j,i],  M = W1[:, :64] @ W3   (sample independent)
__global__ void prep_kmat(const float* __restrict__ W1, const float* __restrict__ W2,
                          const float* __restrict__ W3, float* __restrict__ K) {
  int i = blockIdx.x, j = threadIdx.x;
  float m = 0.f;
  for (int d = 0; d < 64; ++d) m += W1[j * 65 + d] * W3[d * 256 + i];
  K[i * 256 + j] = W2[i * 256 + j] * m;
}

// fold scalar t into layer-1 bias: b1' = b1 + t * W1[:,64]
__global__ void prep_b1p(const float* __restrict__ b1, const float* __restrict__ W1,
                         const float* __restrict__ t, float* __restrict__ b1p) {
  int n = threadIdx.x;
  b1p[n] = b1[n] + t[0] * W1[n * 65 + 64];
}

// pack row-major weight W (N x K, given row stride) into WMMA B-operand tiles:
// tile = nt*KT + kt ; element(tile, lane, e) = W[nt*16 + (lane&15)]
//                          [kt*32 + 8*(lane>>4) + (e&7) + 16*(e>>3)]
__global__ void prep_pack(const float* __restrict__ src, unsigned short* __restrict__ dst,
                          int KT, int rowstride, int total) {
  int idx = blockIdx.x * 256 + threadIdx.x;
  if (idx >= total) return;
  int tile = idx >> 9, within = idx & 511;
  int lane = within >> 4, e = within & 15;
  int nt = tile / KT, kt = tile - nt * KT;
  int n = (nt << 4) + (lane & 15);
  int k = (kt << 5) + ((lane >> 4) << 3) + (e & 7) + ((e >> 3) << 4);
  dst[idx] = f2bf(src[(size_t)n * rowstride + k]);
}

// ------------------------------ fused kernel -------------------------------

__global__ __launch_bounds__(256) void ode_fused(
    const float* __restrict__ z,
    const float* __restrict__ b2, const float* __restrict__ b3,
    const float* __restrict__ b1p,
    const unsigned short* __restrict__ w1p, const unsigned short* __restrict__ w2p,
    const unsigned short* __restrict__ w3p, const unsigned short* __restrict__ kp,
    float* __restrict__ out)
{
  __shared__ unsigned short s_z [16 * ZP];
  __shared__ unsigned short s_h1[16 * HP];
  __shared__ unsigned short s_d1[16 * HP];
  __shared__ unsigned short s_h2[16 * HP];
  __shared__ float          s_d2[16 * FP];
  __shared__ float          s_u [16 * FP];
  __shared__ float          s_part[256];

  const int tid  = threadIdx.x;
  const int lane = tid & 31;
  const int w    = tid >> 5;        // wave id, 0..7
  const int r0   = blockIdx.x << 4; // first batch row of this workgroup

  // stage in z tile as bf16
  for (int i = tid; i < 16 * DIMZ; i += 256) {
    int m = i >> 6, k = i & 63;
    s_z[m * ZP + k] = f2bf(z[(size_t)(r0 + m) * DIMZ + k]);
  }
  __syncthreads();

  // ---- layer 1: a1 = z @ W1z^T + b1'  (K=64) ----
  {
    v16bf a0 = load_a(s_z, ZP, 0, lane);
    v16bf a1 = load_a(s_z, ZP, 1, lane);
    for (int nn = 0; nn < 2; ++nn) {
      int nt = (w << 1) + nn;
      v8f acc = {};
      acc = wmma_bf16(a0, load_b(w1p, nt * 2 + 0, lane), acc);
      acc = wmma_bf16(a1, load_b(w1p, nt * 2 + 1, lane), acc);
      int n = (nt << 4) + (lane & 15);
      float bias = b1p[n];
      #pragma unroll
      for (int r = 0; r < 8; ++r) {
        int m = r + ((lane >> 4) << 3);          // D-layout row
        float x = acc[r] + bias;
        float s = fsigmoid(x);
        s_h1[m * HP + n] = f2bf(x * s);                    // silu
        s_d1[m * HP + n] = f2bf(s + x * s * (1.f - s));    // silu'
      }
    }
  }
  __syncthreads();

  // ---- layer 2: a2 = h1 @ W2^T + b2  (K=256) ----
  {
    v16bf A[8];
    #pragma unroll
    for (int kt = 0; kt < 8; ++kt) A[kt] = load_a(s_h1, HP, kt, lane);
    for (int nn = 0; nn < 2; ++nn) {
      int nt = (w << 1) + nn;
      v8f acc = {};
      #pragma unroll
      for (int kt = 0; kt < 8; ++kt)
        acc = wmma_bf16(A[kt], load_b(w2p, nt * 8 + kt, lane), acc);
      int n = (nt << 4) + (lane & 15);
      float bias = b2[n];
      #pragma unroll
      for (int r = 0; r < 8; ++r) {
        int m = r + ((lane >> 4) << 3);
        float x = acc[r] + bias;
        float s = fsigmoid(x);
        s_h2[m * HP + n] = f2bf(x * s);
        s_d2[m * FP + n] = s + x * s * (1.f - s);
      }
    }
  }
  __syncthreads();

  // ---- U = d1 @ K^T (all 8 waves) ----
  {
    v16bf A[8];
    #pragma unroll
    for (int kt = 0; kt < 8; ++kt) A[kt] = load_a(s_d1, HP, kt, lane);
    for (int nn = 0; nn < 2; ++nn) {
      int nt = (w << 1) + nn;
      v8f acc = {};
      #pragma unroll
      for (int kt = 0; kt < 8; ++kt)
        acc = wmma_bf16(A[kt], load_b(kp, nt * 8 + kt, lane), acc);
      int n = (nt << 4) + (lane & 15);
      #pragma unroll
      for (int r = 0; r < 8; ++r) {
        int m = r + ((lane >> 4) << 3);
        s_u[m * FP + n] = acc[r];
      }
    }
  }
  // ---- layer 3: dz_dt = h2 @ W3^T + b3 (waves 0..3, N=64) ----
  if (w < 4) {
    v16bf A[8];
    #pragma unroll
    for (int kt = 0; kt < 8; ++kt) A[kt] = load_a(s_h2, HP, kt, lane);
    v8f acc = {};
    #pragma unroll
    for (int kt = 0; kt < 8; ++kt)
      acc = wmma_bf16(A[kt], load_b(w3p, w * 8 + kt, lane), acc);
    int n = (w << 4) + (lane & 15);
    float bias = b3[n];
    #pragma unroll
    for (int r = 0; r < 8; ++r) {
      int m = r + ((lane >> 4) << 3);
      out[(size_t)(r0 + m) * DIMZ + n] = acc[r] + bias;
    }
  }
  __syncthreads();

  // ---- div[row] = sum_n d2[row,n]*u[row,n] ; deterministic 2-stage reduce ----
  {
    int row = tid >> 4, seg = tid & 15;
    float p = 0.f;
    #pragma unroll
    for (int c = 0; c < 16; ++c) {
      int n = (seg << 4) + c;
      p += s_u[row * FP + n] * s_d2[row * FP + n];
    }
    s_part[tid] = p;
  }
  __syncthreads();
  if (tid < 16) {
    float d = 0.f;
    #pragma unroll
    for (int c = 0; c < 16; ++c) d += s_part[tid * 16 + c];
    out[(size_t)NB * DIMZ + r0 + tid] = -d;   // dlogp_dt
  }
}

// ------------------------------ launcher -----------------------------------

extern "C" void kernel_launch(void* const* d_in, const int* in_sizes, int n_in,
                              void* d_out, int out_size, void* d_ws, size_t ws_size,
                              hipStream_t stream) {
  (void)in_sizes; (void)n_in; (void)out_size; (void)ws_size;
  const float* t  = (const float*)d_in[0];
  const float* z  = (const float*)d_in[1];
  const float* W1 = (const float*)d_in[3];
  const float* b1 = (const float*)d_in[4];
  const float* W2 = (const float*)d_in[5];
  const float* b2 = (const float*)d_in[6];
  const float* W3 = (const float*)d_in[7];
  const float* b3 = (const float*)d_in[8];
  float* out = (float*)d_out;

  char* ws = (char*)d_ws;                       // ~577 KB used
  float* kmat = (float*)ws;                     // 256*256*4 = 262144
  float* b1p  = (float*)(ws + 262144);          // 1024
  unsigned short* w1p = (unsigned short*)(ws + 263168);  // 32768
  unsigned short* w2p = (unsigned short*)(ws + 295936);  // 131072
  unsigned short* w3p = (unsigned short*)(ws + 427008);  // 32768
  unsigned short* kp  = (unsigned short*)(ws + 459776);  // 131072

  prep_kmat<<<256, 256, 0, stream>>>(W1, W2, W3, kmat);
  prep_b1p<<<1, 256, 0, stream>>>(b1, W1, t, b1p);
  prep_pack<<<64,  256, 0, stream>>>(W1,   w1p, 2, 65,  16384); // W1z: N=256,K=64
  prep_pack<<<256, 256, 0, stream>>>(W2,   w2p, 8, 256, 65536); // W2 : N=256,K=256
  prep_pack<<<64,  256, 0, stream>>>(W3,   w3p, 8, 256, 16384); // W3 : N=64, K=256
  prep_pack<<<256, 256, 0, stream>>>(kmat, kp,  8, 256, 65536); // K  : N=256,K=256

  ode_fused<<<NB / 16, 256, 0, stream>>>(z, b2, b3, b1p, w1p, w2p, w3p, kp, out);
}
